// _CrossAttND_66005057405645
// MI455X (gfx1250) — matrix operand order repaired and checked by reading, not compiled, not run on hardware
//
#include <hip/hip_runtime.h>
#include <hip/hip_bf16.h>
#include <math.h>

#define BATCH 8
#define CIN   256
#define CMASK 64
#define CIMG  192
#define NP    4096   // 64*64 pixels (full res)
#define HWP   1024   // 32*32 pixels (pooled)
#define EPSBN 1e-5f
#define PSTR  1032   // padded row stride (bf16 elems) for P in LDS: 2064B % 256B = 16
#define ATTN_SMEM (32 * PSTR * 2)

typedef __bf16 bf16;
typedef __attribute__((ext_vector_type(16))) __bf16 v16bf;
typedef __attribute__((ext_vector_type(8)))  float  v8f;

union FragU { v16bf v; uint4 q[2]; };

// ---- WMMA fragment loaders (wave32, 16x16x32 bf16 shapes) -------------------
// A (16x32, row-major [M][K]): lanes 0-15 row M=lane hold K 0-7 & 16-23;
// lanes 16-31 hold K 8-15 & 24-31 (ISA 7.12.2 16-bit A table).
__device__ __forceinline__ v16bf load_a_frag(const bf16* A, int lda, int lane) {
  const int r  = lane & 15;
  const int kb = (lane >> 4) << 3;
  FragU f;
  f.q[0] = *reinterpret_cast<const uint4*>(A + (size_t)r * lda + kb);
  f.q[1] = *reinterpret_cast<const uint4*>(A + (size_t)r * lda + kb + 16);
  return f.v;
}
// B (32x16) stored transposed [N][K]: lanes 0-15 col N=lane hold K 0-7 & 16-23,
// lanes 16-31 hold K 8-15 & 24-31 (same K order as A).
__device__ __forceinline__ v16bf load_bt_frag(const bf16* Bt, int ldb, int lane) {
  const int r  = lane & 15;
  const int kb = (lane >> 4) << 3;
  FragU f;
  f.q[0] = *reinterpret_cast<const uint4*>(Bt + (size_t)r * ldb + kb);
  f.q[1] = *reinterpret_cast<const uint4*>(Bt + (size_t)r * ldb + kb + 16);
  return f.v;
}
#define WMMA_BF16(a,b,c) \
  __builtin_amdgcn_wmma_f32_16x16x32_bf16(false,(a),false,(b),(short)0,(c),false,false)

// C/D layout: element r, lane l -> M = r + ((l>>4)<<3), N = (l&15).

// Per-wave 32(M) x 64(N) GEMM, software-pipelined (prefetch next K-step).
// A row-major rows m0..m0+31, B as [N][K]. acc[rt*4+j]: rows rt*16, cols j*16.
template <int K>
__device__ __forceinline__ void gemm_32x64(const bf16* __restrict__ A, int lda,
                                           const bf16* __restrict__ Bt, int ldb,
                                           int lane, v8f acc[8]) {
  v16bf a0 = load_a_frag(A, lda, lane);
  v16bf a1 = load_a_frag(A + (size_t)16 * lda, lda, lane);
  v16bf b0 = load_bt_frag(Bt, ldb, lane);
  v16bf b1 = load_bt_frag(Bt + (size_t)16 * ldb, ldb, lane);
  v16bf b2 = load_bt_frag(Bt + (size_t)32 * ldb, ldb, lane);
  v16bf b3 = load_bt_frag(Bt + (size_t)48 * ldb, ldb, lane);
#pragma unroll
  for (int k0 = 0; k0 < K; k0 += 32) {
    v16bf na0 = a0, na1 = a1, nb0 = b0, nb1 = b1, nb2 = b2, nb3 = b3;
    if (k0 + 32 < K) {  // prefetch next step before issuing wmmas
      na0 = load_a_frag(A + k0 + 32, lda, lane);
      na1 = load_a_frag(A + (size_t)16 * lda + k0 + 32, lda, lane);
      nb0 = load_bt_frag(Bt + k0 + 32, ldb, lane);
      nb1 = load_bt_frag(Bt + (size_t)16 * ldb + k0 + 32, ldb, lane);
      nb2 = load_bt_frag(Bt + (size_t)32 * ldb + k0 + 32, ldb, lane);
      nb3 = load_bt_frag(Bt + (size_t)48 * ldb + k0 + 32, ldb, lane);
    }
    acc[0] = WMMA_BF16(a0, b0, acc[0]);
    acc[1] = WMMA_BF16(a0, b1, acc[1]);
    acc[2] = WMMA_BF16(a0, b2, acc[2]);
    acc[3] = WMMA_BF16(a0, b3, acc[3]);
    acc[4] = WMMA_BF16(a1, b0, acc[4]);
    acc[5] = WMMA_BF16(a1, b1, acc[5]);
    acc[6] = WMMA_BF16(a1, b2, acc[6]);
    acc[7] = WMMA_BF16(a1, b3, acc[7]);
    a0 = na0; a1 = na1; b0 = nb0; b1 = nb1; b2 = nb2; b3 = nb3;
  }
}

// Per-wave 16(M) x 64(N) GEMM (used by phi conv).
template <int K>
__device__ __forceinline__ void gemm_16x64(const bf16* __restrict__ Arow, int lda,
                                           const bf16* __restrict__ Bt, int ldb,
                                           int lane, v8f acc[4]) {
#pragma unroll
  for (int k0 = 0; k0 < K; k0 += 32) {
    v16bf a = load_a_frag(Arow + k0, lda, lane);
#pragma unroll
    for (int j = 0; j < 4; ++j) {
      v16bf bq = load_bt_frag(Bt + (size_t)(j * 16) * ldb + k0, ldb, lane);
      acc[j] = WMMA_BF16(a, bq, acc[j]);
    }
  }
}

// ---- Prep: x (B,C,4096) f32 -> xT (B,4096,C) bf16 (LDS-tiled transpose) -----
__global__ __launch_bounds__(256) void k_xT(const float* __restrict__ x,
                                            bf16* __restrict__ xT) {
  __shared__ float t[32][33];
  const int b = blockIdx.z, pb = blockIdx.x * 32, cb = blockIdx.y * 32;
  const int tx = threadIdx.x & 31, ty = threadIdx.x >> 5;
  const float* xb = x + (size_t)b * CIN * NP;
#pragma unroll
  for (int i = 0; i < 32; i += 8)
    t[ty + i][tx] = xb[(size_t)(cb + ty + i) * NP + pb + tx];
  __syncthreads();
  bf16* o = xT + (size_t)b * NP * CIN;
#pragma unroll
  for (int i = 0; i < 32; i += 8)
    o[(size_t)(pb + ty + i) * CIN + cb + tx] = (bf16)t[tx][ty + i];
}

__global__ __launch_bounds__(256) void k_cvt(const float* __restrict__ in,
                                             bf16* __restrict__ out, int n) {
  int i = blockIdx.x * 256 + threadIdx.x;
  if (i < n) out[i] = (bf16)in[i];
}

// ---- theta conv: thB[b][p][o] = sum_c xT[b][p][c<64] * tw[o][c] + tb[o] -----
__global__ __launch_bounds__(256) void k_theta(const bf16* __restrict__ xT,
                                               const bf16* __restrict__ tw,
                                               const float* __restrict__ tb,
                                               bf16* __restrict__ thB) {
  const int b = blockIdx.y, wv = threadIdx.x >> 5, lane = threadIdx.x & 31;
  const int m0 = blockIdx.x * 256 + wv * 32;
  v8f acc[8] = {};
  gemm_32x64<CMASK>(xT + (size_t)b * NP * CIN + (size_t)m0 * CIN, CIN,
                    tw, CMASK, lane, acc);
  const int half = (lane >> 4) << 3, nl = lane & 15;
  bf16* o = thB + (size_t)b * NP * CMASK;
#pragma unroll
  for (int rt = 0; rt < 2; ++rt)
#pragma unroll
    for (int j = 0; j < 4; ++j)
#pragma unroll
      for (int r = 0; r < 8; ++r) {
        int p = m0 + rt * 16 + r + half, c = j * 16 + nl;
        o[(size_t)p * CMASK + c] = (bf16)(acc[rt * 4 + j][r] + tb[c]);
      }
}

// ---- phi conv + 2x2 maxpool: phT[b][m][o], m = pooled pixel -----------------
__global__ __launch_bounds__(256) void k_phi(const bf16* __restrict__ xT,
                                             const bf16* __restrict__ pw,
                                             const float* __restrict__ pb,
                                             bf16* __restrict__ phT) {
  const int b = blockIdx.z, wv = threadIdx.x >> 5, lane = threadIdx.x & 31;
  const int mblk = blockIdx.x * 128, m0 = mblk + wv * 16;
  v8f acc[4] = {};
  gemm_16x64<CIMG>(xT + (size_t)b * NP * CIN + (size_t)m0 * CIN + CMASK, CIN,
                   pw, CIMG, lane, acc);
  __shared__ float tile[128][65];
  const int half = (lane >> 4) << 3, nl = lane & 15;
#pragma unroll
  for (int j = 0; j < 4; ++j)
#pragma unroll
    for (int r = 0; r < 8; ++r)
      tile[wv * 16 + r + half][j * 16 + nl] = acc[j][r] + pb[j * 16 + nl];
  __syncthreads();
  // block covers image rows 2*blockIdx.x, 2*blockIdx.x+1 -> pooled row blockIdx.x
  bf16* o = phT + (size_t)b * HWP * CMASK + (size_t)blockIdx.x * 32 * CMASK;
  for (int idx = threadIdx.x; idx < 32 * 64; idx += 256) {
    int c = idx & 63, jj = idx >> 6, w0 = jj * 2;
    float v = fmaxf(fmaxf(tile[w0][c], tile[w0 + 1][c]),
                    fmaxf(tile[64 + w0][c], tile[64 + w0 + 1][c]));
    o[(size_t)jj * CMASK + c] = (bf16)v;
  }
}

// ---- g conv + 2x2 maxpool: gT[b][c][m] (channel-major = [N][K] for AV) ------
// Block tile 128(M) x 128(N): waves arranged 4(M) x 2(N), each 32x64.
__global__ __launch_bounds__(256) void k_g(const bf16* __restrict__ xT,
                                           const bf16* __restrict__ gw,
                                           const float* __restrict__ gb,
                                           bf16* __restrict__ gT) {
  const int b = blockIdx.z;
  const int wv = threadIdx.x >> 5, lane = threadIdx.x & 31;
  const int wm = wv >> 1, wc = wv & 1;
  const int m0 = blockIdx.x * 128 + wm * 32;
  const int nb = blockIdx.y * 128 + wc * 64;
  v8f acc[8] = {};
  gemm_32x64<CIN>(xT + (size_t)b * NP * CIN + (size_t)m0 * CIN, CIN,
                  gw + (size_t)nb * CIN, CIN, lane, acc);
  __shared__ float tile[128][65];
  const int half = (lane >> 4) << 3, nl = lane & 15;
#pragma unroll
  for (int pass = 0; pass < 2; ++pass) {
    if (wc == pass) {
#pragma unroll
      for (int rt = 0; rt < 2; ++rt)
#pragma unroll
        for (int j = 0; j < 4; ++j)
#pragma unroll
          for (int r = 0; r < 8; ++r)
            tile[wm * 32 + rt * 16 + r + half][j * 16 + nl] =
                acc[rt * 4 + j][r] + gb[nb + j * 16 + nl];
    }
    __syncthreads();
    const int ncol = blockIdx.y * 128 + pass * 64;
    bf16* o = gT + (size_t)b * CIN * HWP + (size_t)blockIdx.x * 32;
    for (int idx = threadIdx.x; idx < 32 * 64; idx += 256) {
      int jj = idx & 31, c = idx >> 5, w0 = jj * 2;
      float v = fmaxf(fmaxf(tile[w0][c], tile[w0 + 1][c]),
                      fmaxf(tile[64 + w0][c], tile[64 + w0 + 1][c]));
      o[(size_t)(ncol + c) * HWP + jj] = (bf16)v;
    }
    __syncthreads();
  }
}

// ---- Fused attention: scores -> softmax -> y = P @ g ------------------------
// One block = one batch, 32 rows. Scores: 8 waves split the 1024 columns
// (wave w: cols w*64 + {0,512}); B frags shared across both 16-row halves.
// AV: 8 waves split 256 channels (w*32), 32x32 per wave, B double-buffered.
__global__ __launch_bounds__(256) void k_attn(const bf16* __restrict__ thB,
                                              const bf16* __restrict__ phT,
                                              const bf16* __restrict__ gT,
                                              bf16* __restrict__ yB) {
  extern __shared__ __align__(16) bf16 P[];   // [32][PSTR] unnormalized numerators
  __shared__ float red[8][32];
  __shared__ float rowmax[32], rowsum[32];

  const int b = blockIdx.y, m0 = blockIdx.x * 32;
  const int wv = threadIdx.x >> 5, lane = threadIdx.x & 31;
  const int half = (lane >> 4) << 3, nl = lane & 15;

  const bf16* Ath = thB + (size_t)b * NP * CMASK + (size_t)m0 * CMASK;
  v16bf a00 = load_a_frag(Ath, CMASK, lane);                    // rows 0-15, K 0-31
  v16bf a01 = load_a_frag(Ath + 32, CMASK, lane);               // rows 0-15, K 32-63
  v16bf a10 = load_a_frag(Ath + 16 * CMASK, CMASK, lane);       // rows 16-31
  v16bf a11 = load_a_frag(Ath + 16 * CMASK + 32, CMASK, lane);

  const bf16* phib = phT + (size_t)b * HWP * CMASK;
  v8f s[2][8];
#pragma unroll
  for (int q = 0; q < 8; ++q) {
    const int n = wv * 64 + (q >> 2) * 512 + (q & 3) * 16;
    v16bf b0 = load_bt_frag(phib + (size_t)n * CMASK, CMASK, lane);
    v16bf b1 = load_bt_frag(phib + (size_t)n * CMASK + 32, CMASK, lane);
    v8f c0 = {}, c1 = {};
    c0 = WMMA_BF16(a00, b0, c0);
    c0 = WMMA_BF16(a01, b1, c0);
    c1 = WMMA_BF16(a10, b0, c1);
    c1 = WMMA_BF16(a11, b1, c1);
    s[0][q] = c0;
    s[1][q] = c1;
  }

  // row max: per-lane over 8 tiles, shuffle across the 16 lanes of each half,
  // then across the 8 waves through LDS.
  float pmax[2][8];
#pragma unroll
  for (int rt = 0; rt < 2; ++rt)
#pragma unroll
    for (int r = 0; r < 8; ++r) {
      float m = s[rt][0][r];
#pragma unroll
      for (int q = 1; q < 8; ++q) m = fmaxf(m, s[rt][q][r]);
#pragma unroll
      for (int off = 8; off >= 1; off >>= 1) m = fmaxf(m, __shfl_xor(m, off, 16));
      pmax[rt][r] = m;
    }
  if (nl == 0)
#pragma unroll
    for (int rt = 0; rt < 2; ++rt)
#pragma unroll
      for (int r = 0; r < 8; ++r) red[wv][rt * 16 + half + r] = pmax[rt][r];
  __syncthreads();
  if (threadIdx.x < 32) {
    float m = red[0][threadIdx.x];
#pragma unroll
    for (int w = 1; w < 8; ++w) m = fmaxf(m, red[w][threadIdx.x]);
    rowmax[threadIdx.x] = m;
  }
  __syncthreads();

  // exp, stash bf16 P into LDS, accumulate per-row sums
  float psum[2][8];
#pragma unroll
  for (int rt = 0; rt < 2; ++rt)
#pragma unroll
    for (int r = 0; r < 8; ++r) psum[rt][r] = 0.f;
#pragma unroll
  for (int q = 0; q < 8; ++q) {
    const int n = wv * 64 + (q >> 2) * 512 + (q & 3) * 16 + nl;
#pragma unroll
    for (int rt = 0; rt < 2; ++rt)
#pragma unroll
      for (int r = 0; r < 8; ++r) {
        const int row = rt * 16 + r + half;
        float e = __expf(s[rt][q][r] - rowmax[row]);
        psum[rt][r] += e;
        P[(size_t)row * PSTR + n] = (bf16)e;
      }
  }
#pragma unroll
  for (int rt = 0; rt < 2; ++rt)
#pragma unroll
    for (int r = 0; r < 8; ++r) {
      float v = psum[rt][r];
#pragma unroll
      for (int off = 8; off >= 1; off >>= 1) v += __shfl_xor(v, off, 16);
      psum[rt][r] = v;
    }
  if (nl == 0)
#pragma unroll
    for (int rt = 0; rt < 2; ++rt)
#pragma unroll
      for (int r = 0; r < 8; ++r) red[wv][rt * 16 + half + r] = psum[rt][r];
  __syncthreads();
  if (threadIdx.x < 32) {
    float v = 0.f;
#pragma unroll
    for (int w = 0; w < 8; ++w) v += red[w][threadIdx.x];
    rowsum[threadIdx.x] = v;
  }
  __syncthreads();

  // y(32x32 per wave) = P(32x1024) @ g^T(256x1024 as [N][K]); K = 1024.
  const bf16* gbp = gT + (size_t)b * CIN * HWP;
  const int n0 = wv * 32;
  v8f y[4] = {};
  v16bf b0 = load_bt_frag(gbp + (size_t)n0 * HWP, HWP, lane);
  v16bf b1 = load_bt_frag(gbp + (size_t)(n0 + 16) * HWP, HWP, lane);
#pragma unroll 4
  for (int k0 = 0; k0 < HWP; k0 += 32) {
    v16bf nb0 = b0, nb1 = b1;
    if (k0 + 32 < HWP) {  // prefetch next B frags (global)
      nb0 = load_bt_frag(gbp + (size_t)n0 * HWP + k0 + 32, HWP, lane);
      nb1 = load_bt_frag(gbp + (size_t)(n0 + 16) * HWP + k0 + 32, HWP, lane);
    }
    v16bf pa0 = load_a_frag(P + k0, PSTR, lane);              // rows 0-15 (LDS)
    v16bf pa1 = load_a_frag(P + (size_t)16 * PSTR + k0, PSTR, lane);
    y[0] = WMMA_BF16(pa0, b0, y[0]);
    y[1] = WMMA_BF16(pa0, b1, y[1]);
    y[2] = WMMA_BF16(pa1, b0, y[2]);
    y[3] = WMMA_BF16(pa1, b1, y[3]);
    b0 = nb0; b1 = nb1;
  }
  bf16* yo = yB + (size_t)b * NP * CIN + (size_t)m0 * CIN;
#pragma unroll
  for (int rt = 0; rt < 2; ++rt)
#pragma unroll
    for (int r = 0; r < 8; ++r) {
      const int row = rt * 16 + r + half;
      const float sc = 1.0f / rowsum[row];
      yo[(size_t)row * CIN + n0 + nl]      = (bf16)(y[rt * 2 + 0][r] * sc);
      yo[(size_t)row * CIN + n0 + 16 + nl] = (bf16)(y[rt * 2 + 1][r] * sc);
    }
}

// ---- W conv + batchnorm + residual, coalesced fp32 output via LDS -----------
// Block tile 128(M) x 128(N): waves 4(M) x 2(N), each 32x64; 2-pass epilogue.
__global__ __launch_bounds__(256) void k_out(const bf16* __restrict__ yB,
                                             const bf16* __restrict__ Ww,
                                             const float* __restrict__ Wb,
                                             const float* __restrict__ gamma,
                                             const float* __restrict__ beta,
                                             const float* __restrict__ mean,
                                             const float* __restrict__ var,
                                             const float* __restrict__ x,
                                             float* __restrict__ out) {
  const int b = blockIdx.z;
  const int wv = threadIdx.x >> 5, lane = threadIdx.x & 31;
  const int wm = wv >> 1, wc = wv & 1;
  const int mblk = blockIdx.x * 128, m0 = mblk + wm * 32;
  const int nb = blockIdx.y * 128 + wc * 64;
  v8f acc[8] = {};
  gemm_32x64<CIN>(yB + (size_t)b * NP * CIN + (size_t)m0 * CIN, CIN,
                  Ww + (size_t)nb * CIN, CIN, lane, acc);
  __shared__ float tile[128][65];
  const int half = (lane >> 4) << 3, nl = lane & 15;
#pragma unroll
  for (int pass = 0; pass < 2; ++pass) {
    if (wc == pass) {
#pragma unroll
      for (int rt = 0; rt < 2; ++rt)
#pragma unroll
        for (int j = 0; j < 4; ++j)
#pragma unroll
          for (int r = 0; r < 8; ++r)
            tile[wm * 32 + rt * 16 + r + half][j * 16 + nl] =
                acc[rt * 4 + j][r] + Wb[nb + j * 16 + nl];
    }
    __syncthreads();
    const int ncol = blockIdx.y * 128 + pass * 64;
    for (int idx = threadIdx.x; idx < 128 * 64; idx += 256) {
      const int pl = idx & 127, cl = idx >> 7;
      const int c = ncol + cl, p = mblk + pl;
      const size_t gi = (size_t)b * CIN * NP + (size_t)c * NP + p;
      float v = tile[pl][cl];
      v = (v - mean[c]) * (rsqrtf(var[c] + EPSBN) * gamma[c]) + beta[c];
      out[gi] = v + x[gi];
    }
    __syncthreads();
  }
}

extern "C" void kernel_launch(void* const* d_in, const int* in_sizes, int n_in,
                              void* d_out, int out_size, void* d_ws, size_t ws_size,
                              hipStream_t stream) {
  (void)in_sizes; (void)n_in; (void)out_size; (void)ws_size;
  const float* x     = (const float*)d_in[0];
  const float* g_w   = (const float*)d_in[1];
  const float* g_b   = (const float*)d_in[2];
  const float* th_w  = (const float*)d_in[3];
  const float* th_b  = (const float*)d_in[4];
  const float* ph_w  = (const float*)d_in[5];
  const float* ph_b  = (const float*)d_in[6];
  const float* W_w   = (const float*)d_in[7];
  const float* W_b   = (const float*)d_in[8];
  const float* gamma = (const float*)d_in[9];
  const float* beta  = (const float*)d_in[10];
  const float* mean  = (const float*)d_in[11];
  const float* var   = (const float*)d_in[12];
  float* out = (float*)d_out;

  char* ws = (char*)d_ws;
  size_t off = 0;
  auto alloc = [&](size_t bytes) {
    void* p = ws + off;
    off += (bytes + 255) & ~(size_t)255;
    return p;
  };
  bf16* xT  = (bf16*)alloc((size_t)BATCH * NP * CIN * 2);    // 16.8 MB
  bf16* gT  = (bf16*)alloc((size_t)BATCH * CIN * HWP * 2);   //  4.2 MB
  bf16* thB = (bf16*)alloc((size_t)BATCH * NP * CMASK * 2);  //  4.2 MB
  bf16* phT = (bf16*)alloc((size_t)BATCH * HWP * CMASK * 2); //  1.0 MB
  bf16* yB  = (bf16*)alloc((size_t)BATCH * NP * CIN * 2);    // 16.8 MB
  bf16* gwB = (bf16*)alloc((size_t)CIN * CIN * 2);
  bf16* twB = (bf16*)alloc((size_t)CMASK * CMASK * 2);
  bf16* pwB = (bf16*)alloc((size_t)CMASK * CIMG * 2);
  bf16* WwB = (bf16*)alloc((size_t)CIN * CIN * 2);

  (void)hipFuncSetAttribute((const void*)k_attn,
                            hipFuncAttributeMaxDynamicSharedMemorySize,
                            ATTN_SMEM);

  k_xT<<<dim3(NP / 32, CIN / 32, BATCH), 256, 0, stream>>>(x, xT);
  k_cvt<<<(CIN * CIN + 255) / 256, 256, 0, stream>>>(g_w, gwB, CIN * CIN);
  k_cvt<<<(CMASK * CMASK + 255) / 256, 256, 0, stream>>>(th_w, twB, CMASK * CMASK);
  k_cvt<<<(CMASK * CIMG + 255) / 256, 256, 0, stream>>>(ph_w, pwB, CMASK * CIMG);
  k_cvt<<<(CIN * CIN + 255) / 256, 256, 0, stream>>>(W_w, WwB, CIN * CIN);

  k_theta<<<dim3(NP / 256, BATCH), 256, 0, stream>>>(xT, twB, th_b, thB);
  k_phi<<<dim3(NP / 128, 1, BATCH), 256, 0, stream>>>(xT, pwB, ph_b, phT);
  k_g<<<dim3(NP / 128, CIN / 128, BATCH), 256, 0, stream>>>(xT, gwB, g_b, gT);

  k_attn<<<dim3(NP / 32, BATCH), 256, ATTN_SMEM, stream>>>(thB, phT, gT, yB);

  k_out<<<dim3(NP / 128, CIN / 128, BATCH), 256, 0, stream>>>(
      yB, WwB, W_b, gamma, beta, mean, var, x, out);
}
